// SenderGRULM_55061480734837
// MI455X (gfx1250) — matrix operand order repaired and checked by reading, not compiled, not run on hardware
//
#include <hip/hip_runtime.h>
#include <hip/hip_bf16.h>

// Problem constants (match reference)
#define B_   128
#define V_   32000
#define D_   512
#define E_   256
#define MAXLEN 24
#define STEPS  22
#define NWG_GEMM 500          // V_/64 column strips for the big GEMM
#define KT_OUT (D_ / 32)      // 16 K-tiles in the output GEMM

typedef __attribute__((ext_vector_type(16))) __bf16 v16bf;
typedef __attribute__((ext_vector_type(8)))  float  v8f;
typedef __attribute__((ext_vector_type(4)))  int    v4i;

union Frag { v16bf v; float4 f[2]; };

#ifndef __has_builtin
#define __has_builtin(x) 0
#endif

// gfx1250 async global->LDS copy (ASYNCcnt path), with synchronous fallback.
#if defined(__HIP_DEVICE_COMPILE__) && __has_builtin(__builtin_amdgcn_global_load_async_to_lds_b128)
#define HAVE_ASYNC_LDS 1
#else
#define HAVE_ASYNC_LDS 0
#endif

__device__ __forceinline__ void copy_b128_to_lds(const __bf16* src, __bf16* dst) {
#if HAVE_ASYNC_LDS
  __builtin_amdgcn_global_load_async_to_lds_b128(
      (__attribute__((address_space(1))) v4i*)src,
      (__attribute__((address_space(3))) v4i*)dst, 0, 0);
#else
  *(float4*)dst = *(const float4*)src;
#endif
}

__device__ __forceinline__ void wait_async0() {
#if HAVE_ASYNC_LDS
#if __has_builtin(__builtin_amdgcn_s_wait_asynccnt)
  __builtin_amdgcn_s_wait_asynccnt(0);
#else
  asm volatile("s_wait_asynccnt 0" ::: "memory");
#endif
#endif
}

// ---------------------------------------------------------------------------
// Utility: order-preserving float -> u32 map (for packed u64 argmax)
// ---------------------------------------------------------------------------
__device__ __forceinline__ unsigned fmap(float f) {
  unsigned u = __float_as_uint(f);
  return (u & 0x80000000u) ? ~u : (u | 0x80000000u);
}

// ---------------------------------------------------------------------------
// WMMA core (direct-load version, used by the small GRU GEMM):
// one wave computes a 32x32 output patch.
//   A: row-major bf16 [M x lda]
//   Bpk: fragment-packed bf16 weights: pk[((nt*KT + kt)*32 + lane)*16 + i]
//        with n = nt*16 + (lane&15), k = kt*32 + (lane>>4)*8 + (i&7) + (i>>3)*16
//        (matches ISA 16-bit B layout: lanes 0-15 K=0..7,16..23, lanes 16-31 K=8..15,24..31)
// A fragments follow the ISA 16-bit A layout -> two contiguous 16B loads/lane.
// ---------------------------------------------------------------------------
template <int KT>
__device__ __forceinline__ void gemm_patch(const __bf16* __restrict__ A, int lda,
                                           const __bf16* __restrict__ Bpk,
                                           int nt0, int m0, int lane,
                                           v8f c[2][2]) {
  for (int kt = 0; kt < KT; ++kt) {
    Frag a[2], b[2];
    const int kc0 = kt * 32 + ((lane >> 4) * 8);
    const int kc1 = kc0 + 16;
#pragma unroll
    for (int i = 0; i < 2; ++i) {
      const __bf16* arow = A + (size_t)(m0 + i * 16 + (lane & 15)) * lda;
      a[i].f[0] = *(const float4*)(arow + kc0);
      a[i].f[1] = *(const float4*)(arow + kc1);
    }
#pragma unroll
    for (int j = 0; j < 2; ++j) {
      const __bf16* bp = Bpk + ((size_t)((nt0 + j) * KT + kt) * 32 + lane) * 16;
      b[j].f[0] = *(const float4*)(bp);
      b[j].f[1] = *(const float4*)(bp + 8);
    }
#pragma unroll
    for (int i = 0; i < 2; ++i)
#pragma unroll
      for (int j = 0; j < 2; ++j)
        c[i][j] = __builtin_amdgcn_wmma_f32_16x16x32_bf16(
            false, a[i].v, false, b[j].v, (short)0, c[i][j], false, false);
  }
}

__device__ __forceinline__ void store_patch(const v8f c[2][2], float* __restrict__ out,
                                            int ldo, int m0, int n0, int lane) {
#pragma unroll
  for (int i = 0; i < 2; ++i)
#pragma unroll
    for (int j = 0; j < 2; ++j)
#pragma unroll
      for (int r = 0; r < 8; ++r)
        out[(size_t)(m0 + i * 16 + ((lane >> 4) * 8) + r) * ldo +
            (n0 + j * 16 + (lane & 15))] = c[i][j][r];
}

// ---------------------------------------------------------------------------
// Prep kernels
// ---------------------------------------------------------------------------
__global__ void k_zero_out(float4* __restrict__ p, size_t n4) {
  size_t i = (size_t)blockIdx.x * blockDim.x + threadIdx.x;
  size_t stride = (size_t)gridDim.x * blockDim.x;
  float4 z = make_float4(0.f, 0.f, 0.f, 0.f);
  for (; i < n4; i += stride) p[i] = z;
}

__global__ void k_set_sos_eos(float* __restrict__ out) {
  int b = threadIdx.x;  // 128 threads
  out[((size_t)b * MAXLEN + 0) * V_ + 0] = 1.0f;                 // SOS
  out[((size_t)b * MAXLEN + (MAXLEN - 1)) * V_ + 1] = 1.0f;      // EOS
}

// fp32 row-major [K x N] weights -> bf16 fragment-packed tiles
__global__ void k_pack(const float* __restrict__ W, __bf16* __restrict__ pk,
                       int K, int N) {
  int idx = blockIdx.x * 256 + threadIdx.x;
  if (idx >= K * N) return;
  const int KT = K >> 5;
  int tile = idx >> 9;
  int lane = (idx >> 4) & 31;
  int i = idx & 15;
  int nt = tile / KT;
  int kt = tile - nt * KT;
  int n = nt * 16 + (lane & 15);
  int k = kt * 32 + ((lane >> 4) * 8) + (i & 7) + ((i >> 3) * 16);
  pk[idx] = (__bf16)W[(size_t)k * N + n];
}

// h0 = concat(proto0, proto1) @ W_init + b_init
__global__ void k_init_h(const float* __restrict__ p0, const float* __restrict__ p1,
                         const float* __restrict__ Wi, const float* __restrict__ bi,
                         float* __restrict__ hf, __bf16* __restrict__ hb) {
  int idx = blockIdx.x * 256 + threadIdx.x;   // 128*512 threads
  int b = idx >> 9, d = idx & 511;
  float acc = bi[d];
  for (int k = 0; k < 512; ++k) acc += p0[b * 512 + k] * Wi[(size_t)k * 512 + d];
  for (int k = 0; k < 512; ++k) acc += p1[b * 512 + k] * Wi[(size_t)(k + 512) * 512 + d];
  hf[idx] = acc;
  hb[idx] = (__bf16)acc;
}

// x0 = emb[SOS=0] broadcast to all batch rows (bf16)
__global__ void k_init_x(const float* __restrict__ emb, __bf16* __restrict__ xb) {
  int b = blockIdx.x, e = threadIdx.x;   // 128 x 256
  xb[b * E_ + e] = (__bf16)emb[e];
}

// ---------------------------------------------------------------------------
// Per-step: GRU gate GEMMs  gi = x@W_ih, gh = h@W_hh   (biases added later)
// grid 24 (1536/64 columns), block 256 (8 waves)
// ---------------------------------------------------------------------------
__global__ void k_gru_gemm(const __bf16* __restrict__ xb, const __bf16* __restrict__ hb,
                           const __bf16* __restrict__ wih_pk, const __bf16* __restrict__ whh_pk,
                           float* __restrict__ gi, float* __restrict__ gh) {
  int tid = threadIdx.x, lane = tid & 31, w = tid >> 5;
  int m0 = (w >> 1) * 32;
  int n0 = blockIdx.x * 64 + (w & 1) * 32;
  int nt0 = n0 >> 4;
  {
    v8f c[2][2] = {};
    gemm_patch<E_ / 32>(xb, E_, wih_pk, nt0, m0, lane, c);
    store_patch(c, gi, 3 * D_, m0, n0, lane);
  }
  {
    v8f c[2][2] = {};
    gemm_patch<D_ / 32>(hb, D_, whh_pk, nt0, m0, lane, c);
    store_patch(c, gh, 3 * D_, m0, n0, lane);
  }
}

// Elementwise GRU gate combine -> h_new (fp32 + bf16 copies)
__global__ void k_gru_elem(const float* __restrict__ gi, const float* __restrict__ gh,
                           const float* __restrict__ b_ih, const float* __restrict__ b_hh,
                           float* __restrict__ hf, __bf16* __restrict__ hb) {
  int idx = blockIdx.x * 256 + threadIdx.x;   // 128*512
  int b = idx >> 9, d = idx & 511;
  size_t row = (size_t)b * (3 * D_);
  float ir = gi[row + d]            + b_ih[d];
  float iz = gi[row + D_ + d]       + b_ih[D_ + d];
  float in_ = gi[row + 2 * D_ + d]  + b_ih[2 * D_ + d];
  float hr = gh[row + d]            + b_hh[d];
  float hz = gh[row + D_ + d]       + b_hh[D_ + d];
  float hn = gh[row + 2 * D_ + d]   + b_hh[2 * D_ + d];
  float r = 1.0f / (1.0f + __expf(-(ir + hr)));
  float z = 1.0f / (1.0f + __expf(-(iz + hz)));
  float n = tanhf(in_ + r * hn);
  float h = hf[idx];
  float hnew = (1.0f - z) * n + z * h;
  hf[idx] = hnew;
  hb[idx] = (__bf16)hnew;
}

// ---------------------------------------------------------------------------
// Per-step: logits = h@W_out + b_out + gumbel, fused per-row argmax.
// grid 500 (N strips of 64), block 256 (8 waves). WG covers M=128 x N=64.
// B (W_out) k-slices are staged to LDS via async DMA (ASYNCcnt), double-
// buffered so the k+1 copy overlaps the k-th set of WMMAs. A (h, bf16) is
// loaded directly (tiny, stays hot in WGP$).
// ---------------------------------------------------------------------------
__global__ void k_gemm_argmax(const __bf16* __restrict__ hb, const __bf16* __restrict__ wout_pk,
                              const float* __restrict__ bout, const float* __restrict__ gum,
                              unsigned long long* __restrict__ cand) {
  __shared__ __align__(16) __bf16 ldsB[2][4 * 512];   // 2 x 4KB: 4 fragment tiles (N=64, K=32)
  __shared__ unsigned long long best[B_];

  const int tid = threadIdx.x, lane = tid & 31, w = tid >> 5;
  if (tid < B_) best[tid] = 0ull;

  const int m0 = (w >> 1) * 32;
  const int n0 = blockIdx.x * 64 + (w & 1) * 32;
  const int nt0w = blockIdx.x * 4;          // first of the 4 B tiles this WG uses
  const int nA = n0 + (lane & 15);
  const int nB = nA + 16;

  // Prefetch this lane's gumbel epilogue rows (HBM stream) so it overlaps GEMM.
#pragma unroll
  for (int i = 0; i < 2; ++i)
#pragma unroll
    for (int r = 0; r < 8; ++r)
      __builtin_prefetch(&gum[(size_t)(m0 + i * 16 + ((lane >> 4) * 8) + r) * V_ + nA], 0, 0);

  // Stage one 4KB B k-slice: 256 threads x 16B. Tile tt = tid>>6, 16B chunk tid&63.
  auto stage = [&](int kt, int buf) {
    const __bf16* src = wout_pk +
        ((size_t)((nt0w + (tid >> 6)) * KT_OUT + kt) * 512) + (size_t)(tid & 63) * 8;
    __bf16* dst = &ldsB[buf][(tid >> 6) * 512 + (tid & 63) * 8];
    copy_b128_to_lds(src, dst);
  };

  stage(0, 0);

  v8f c[2][2] = {};
  for (int kt = 0; kt < KT_OUT; ++kt) {
    wait_async0();        // own async staging for slice kt done
    __syncthreads();      // all waves' staging visible; prev reads of other buf done
    if (kt + 1 < KT_OUT) stage(kt + 1, (kt + 1) & 1);

    Frag a[2], b[2];
    const int kc0 = kt * 32 + ((lane >> 4) * 8);
    const int kc1 = kc0 + 16;
#pragma unroll
    for (int i = 0; i < 2; ++i) {
      const __bf16* arow = hb + (size_t)(m0 + i * 16 + (lane & 15)) * D_;
      a[i].f[0] = *(const float4*)(arow + kc0);
      a[i].f[1] = *(const float4*)(arow + kc1);
    }
#pragma unroll
    for (int j = 0; j < 2; ++j) {
      const __bf16* bp = &ldsB[kt & 1][((w & 1) * 2 + j) * 512 + lane * 16];
      b[j].f[0] = *(const float4*)(bp);
      b[j].f[1] = *(const float4*)(bp + 8);
    }
#pragma unroll
    for (int i = 0; i < 2; ++i)
#pragma unroll
      for (int j = 0; j < 2; ++j)
        c[i][j] = __builtin_amdgcn_wmma_f32_16x16x32_bf16(
            false, a[i].v, false, b[j].v, (short)0, c[i][j], false, false);
  }

  const float bo0 = bout[nA];
  const float bo1 = bout[nB];

#pragma unroll
  for (int i = 0; i < 2; ++i) {
    int mbase = m0 + i * 16 + ((lane >> 4) * 8);
#pragma unroll
    for (int r = 0; r < 8; ++r) {
      int m = mbase + r;
      float v0 = c[i][0][r] + bo0 + gum[(size_t)m * V_ + nA];
      float v1 = c[i][1][r] + bo1 + gum[(size_t)m * V_ + nB];
      float v; int n;
      if (v1 > v0) { v = v1; n = nB; } else { v = v0; n = nA; }
      unsigned long long p = ((unsigned long long)fmap(v) << 32) | (unsigned)n;
      // reduce across the 16 lanes that share this row set (stay within half)
#pragma unroll
      for (int s = 1; s < 16; s <<= 1) {
        unsigned long long q = __shfl_xor(p, s, 32);
        if (q > p) p = q;
      }
      if ((lane & 15) == 0) atomicMax(&best[m], p);
    }
  }
  __syncthreads();
  if (tid < B_) cand[(size_t)blockIdx.x * B_ + tid] = best[tid];
}

// Reduce candidates, emit one-hot token into lang, gather emb row -> next x.
__global__ void k_finalize(const unsigned long long* __restrict__ cand,
                           float* __restrict__ out, const float* __restrict__ emb,
                           __bf16* __restrict__ xb, int step) {
  __shared__ unsigned long long red[128];
  int b = blockIdx.x, tid = threadIdx.x;
  unsigned long long bestv = 0ull;
  for (int j = tid; j < NWG_GEMM; j += 128) {
    unsigned long long v = cand[(size_t)j * B_ + b];
    if (v > bestv) bestv = v;
  }
  red[tid] = bestv;
  __syncthreads();
  for (int s = 64; s > 0; s >>= 1) {
    if (tid < s && red[tid + s] > red[tid]) red[tid] = red[tid + s];
    __syncthreads();
  }
  int tok = (int)(red[0] & 0xffffffffull);
  if (tid == 0) out[((size_t)b * MAXLEN + (step + 1)) * V_ + tok] = 1.0f;
  for (int e = tid; e < E_; e += 128) xb[b * E_ + e] = (__bf16)emb[(size_t)tok * E_ + e];
}

// ---------------------------------------------------------------------------
// Launch
// ---------------------------------------------------------------------------
extern "C" void kernel_launch(void* const* d_in, const int* in_sizes, int n_in,
                              void* d_out, int out_size, void* d_ws, size_t ws_size,
                              hipStream_t stream) {
  const float* proto0 = (const float*)d_in[0];
  const float* proto1 = (const float*)d_in[1];
  const float* W_init = (const float*)d_in[2];
  const float* b_init = (const float*)d_in[3];
  const float* W_ih   = (const float*)d_in[4];
  const float* W_hh   = (const float*)d_in[5];
  const float* b_ih   = (const float*)d_in[6];
  const float* b_hh   = (const float*)d_in[7];
  const float* W_out  = (const float*)d_in[8];
  const float* b_out  = (const float*)d_in[9];
  const float* emb    = (const float*)d_in[10];
  const float* gumbel = (const float*)d_in[11];
  float* out = (float*)d_out;

  // Workspace layout (~37 MB, 256B aligned)
  char* base = (char*)d_ws;
  size_t o = 0;
  auto take = [&](size_t bytes) { char* p = base + o; o = (o + bytes + 255) & ~(size_t)255; return p; };
  __bf16* wout_pk = (__bf16*)take((size_t)D_ * V_ * 2);
  __bf16* wih_pk  = (__bf16*)take((size_t)E_ * 3 * D_ * 2);
  __bf16* whh_pk  = (__bf16*)take((size_t)D_ * 3 * D_ * 2);
  float*  hf      = (float*)take((size_t)B_ * D_ * 4);
  __bf16* hb      = (__bf16*)take((size_t)B_ * D_ * 2);
  __bf16* xb      = (__bf16*)take((size_t)B_ * E_ * 2);
  float*  gi      = (float*)take((size_t)B_ * 3 * D_ * 4);
  float*  gh      = (float*)take((size_t)B_ * 3 * D_ * 4);
  unsigned long long* cand = (unsigned long long*)take((size_t)NWG_GEMM * B_ * 8);
  (void)ws_size; (void)in_sizes; (void)n_in;

  // --- prep ---
  size_t n4 = (size_t)out_size / 4;
  k_zero_out<<<8192, 256, 0, stream>>>((float4*)out, n4);
  k_set_sos_eos<<<1, 128, 0, stream>>>(out);
  k_pack<<<((D_ * V_) + 255) / 256, 256, 0, stream>>>(W_out, wout_pk, D_, V_);
  k_pack<<<((E_ * 3 * D_) + 255) / 256, 256, 0, stream>>>(W_ih, wih_pk, E_, 3 * D_);
  k_pack<<<((D_ * 3 * D_) + 255) / 256, 256, 0, stream>>>(W_hh, whh_pk, D_, 3 * D_);
  k_init_h<<<(B_ * D_) / 256, 256, 0, stream>>>(proto0, proto1, W_init, b_init, hf, hb);
  k_init_x<<<B_, E_, 0, stream>>>(emb, xb);

  // --- autoregressive steps ---
  for (int t = 0; t < STEPS; ++t) {
    k_gru_gemm<<<(3 * D_) / 64, 256, 0, stream>>>(xb, hb, wih_pk, whh_pk, gi, gh);
    k_gru_elem<<<(B_ * D_) / 256, 256, 0, stream>>>(gi, gh, b_ih, b_hh, hf, hb);
    k_gemm_argmax<<<NWG_GEMM, 256, 0, stream>>>(hb, wout_pk, b_out,
                                                gumbel + (size_t)t * B_ * V_, cand);
    k_finalize<<<B_, 128, 0, stream>>>(cand, out, emb, xb, t);
  }
}
